// SpectralGradients_14559939133811
// MI455X (gfx1250) — matrix operand order repaired and checked by reading, not compiled, not run on hardware
//
#include <hip/hip_runtime.h>
#include <hip/hip_bf16.h>
#include <math.h>

typedef __attribute__((ext_vector_type(16))) __bf16 bf16x16;
typedef __attribute__((ext_vector_type(8)))  __bf16 bf16x8;
typedef __attribute__((ext_vector_type(8)))  float  f32x8;

#define T_LEN   3000
#define T_EXT   3030
#define T_PAD   3008
#define YSTRIDE 3072
#define NF      97
#define NB      64
#define M_ROWS  6272     /* 64 x-rows + 97*64 filtered rows */
#define H_DIM   256
#define KB1     94       /* 3008/32 k-blocks for GEMM1 */
#define KB2     8        /* 256/32 k-blocks for GEMM2 */

// ---------------------------------------------------------------------------
// Stage 1: filtfilt (forward+backward order-4 IIR), one thread per (filter,batch)
// ---------------------------------------------------------------------------
__global__ __launch_bounds__(64)
void k_filtfilt(const float* __restrict__ x, const float* __restrict__ b_all,
                const float* __restrict__ a_all, const float* __restrict__ zi_all,
                float* __restrict__ yext, float* __restrict__ bs)
{
    const int f = blockIdx.x;
    const int b = threadIdx.x;
    const float* xr = x + (size_t)b * T_LEN;

    const float b0 = b_all[f*5+0], b1c = b_all[f*5+1], b2 = b_all[f*5+2],
                b3 = b_all[f*5+3], b4  = b_all[f*5+4];
    const float a1 = a_all[f*5+1], a2 = a_all[f*5+2],
                a3 = a_all[f*5+3], a4 = a_all[f*5+4];
    const float zi0 = zi_all[f*4+0], zi1 = zi_all[f*4+1],
                zi2 = zi_all[f*4+2], zi3 = zi_all[f*4+3];

    const float x0 = xr[0], xl = xr[T_LEN-1];
    float* ye = yext + (size_t)(f*NB + b) * YSTRIDE;

    // forward pass over extended signal [left(15), x(3000), right(15)]
    const float e0 = 2.f*x0 - xr[15];
    float z0 = zi0*e0, z1 = zi1*e0, z2 = zi2*e0, z3 = zi3*e0;
    float lasty = 0.f;
    for (int t = 0; t < T_EXT; ++t) {
        float xt;
        if (t < 15)        xt = 2.f*x0 - xr[15 - t];
        else if (t < 3015) xt = xr[t - 15];
        else               xt = 2.f*xl - xr[6013 - t];
        float y = fmaf(b0, xt, z0);
        z0 = fmaf(-a1, y, fmaf(b1c, xt, z1));
        z1 = fmaf(-a2, y, fmaf(b2,  xt, z2));
        z2 = fmaf(-a3, y, fmaf(b3,  xt, z3));
        z3 = fmaf(-a4, y, b4*xt);
        ye[t] = y;
        lasty = y;
    }
    // backward pass over flipped forward output; emit trimmed, re-flipped result
    z0 = zi0*lasty; z1 = zi1*lasty; z2 = zi2*lasty; z3 = zi3*lasty;
    float* bsr = bs + (size_t)(f*NB + b) * T_LEN;
    for (int t = 0; t < T_EXT; ++t) {
        float u = ye[T_EXT - 1 - t];
        float y = fmaf(b0, u, z0);
        z0 = fmaf(-a1, y, fmaf(b1c, u, z1));
        z1 = fmaf(-a2, y, fmaf(b2,  u, z2));
        z2 = fmaf(-a3, y, fmaf(b3,  u, z3));
        z3 = fmaf(-a4, y, b4*u);
        int s = 3014 - t;
        if ((unsigned)s < (unsigned)T_LEN) bsr[s] = y;
    }
}

// ---------------------------------------------------------------------------
// Stage 2: bf16 hi/lo packing (split-precision: v = hi + lo, both bf16)
// ---------------------------------------------------------------------------
__global__ __launch_bounds__(256)
void k_packP(const float* __restrict__ x, const float* __restrict__ bs,
             __bf16* __restrict__ Phi, __bf16* __restrict__ Plo)
{
    const int m = blockIdx.y;
    const int k = blockIdx.x*256 + threadIdx.x;
    if (k >= T_PAD) return;
    float v = 0.f;
    if (k < T_LEN)
        v = (m < NB) ? x[(size_t)m*T_LEN + k] : bs[(size_t)(m-NB)*T_LEN + k];
    const size_t o = (size_t)m*T_PAD + k;
    __bf16 hi = (__bf16)v;
    Phi[o] = hi;
    Plo[o] = (__bf16)(v - (float)hi);
}

// W1 packed for B-fragments of GEMM1: layout [kblock(94)][n(256)][kin(32)]
__global__ __launch_bounds__(256)
void k_packW(const float* __restrict__ W1, __bf16* __restrict__ Whi, __bf16* __restrict__ Wlo)
{
    const int gid = blockIdx.x*256 + threadIdx.x;
    if (gid >= KB1*8192) return;
    const int kb  = gid >> 13;
    const int rem = gid & 8191;
    const int n   = rem >> 5;
    const int kin = rem & 31;
    const int tt  = kb*32 + kin;
    const float v = (tt < T_LEN) ? W1[(size_t)tt*H_DIM + n] : 0.f;
    __bf16 hi = (__bf16)v;
    Whi[gid] = hi;
    Wlo[gid] = (__bf16)(v - (float)hi);
}

// W1^T packed for B-fragments of GEMM2: layout [hblock(8)][t(3008)][hin(32)]
__global__ __launch_bounds__(256)
void k_packWT(const float* __restrict__ W1, __bf16* __restrict__ WThi, __bf16* __restrict__ WTlo)
{
    const int gid = blockIdx.x*256 + threadIdx.x;
    if (gid >= KB2*T_PAD*32) return;
    const int hb  = gid / (T_PAD*32);
    const int rem = gid % (T_PAD*32);
    const int t   = rem >> 5;
    const int hin = rem & 31;
    const int h   = hb*32 + hin;
    const float v = (t < T_LEN) ? W1[(size_t)t*H_DIM + h] : 0.f;
    __bf16 hi = (__bf16)v;
    WThi[gid] = hi;
    WTlo[gid] = (__bf16)(v - (float)hi);
}

// G(0) row: g0[t] = sum_h W1[t,h] * w2[h] * (1 - tanh(b1[h])^2)
__global__ __launch_bounds__(256)
void k_g0(const float* __restrict__ W1, const float* __restrict__ bias,
          const float* __restrict__ w2, float* __restrict__ g0)
{
    const int t = blockIdx.x*256 + threadIdx.x;
    if (t >= T_PAD) return;
    if (t >= T_LEN) { g0[t] = 0.f; return; }
    float acc = 0.f;
    for (int h = 0; h < H_DIM; ++h) {
        float th = tanhf(bias[h]);
        acc = fmaf(W1[(size_t)t*H_DIM + h], w2[h]*(1.f - th*th), acc);
    }
    g0[t] = acc;
}

// ---------------------------------------------------------------------------
// WMMA helpers (wave32, v_wmma_f32_16x16x32_bf16, documented fragment layouts)
// ---------------------------------------------------------------------------
static __device__ __forceinline__ bf16x16 ld_frag(const __bf16* p0, const __bf16* p1)
{
    bf16x8 a = *(const bf16x8*)p0;
    bf16x8 b = *(const bf16x8*)p1;
    bf16x16 r;
#pragma unroll
    for (int i = 0; i < 8; ++i) { r[i] = a[i]; r[8+i] = b[i]; }
    return r;
}

#define WMMA_BF16(A, B, C) \
    __builtin_amdgcn_wmma_f32_16x16x32_bf16(false, (A), false, (B), (short)0, (C), false, false)

// GEMM1: Z = P @ W1 (bf16x3), fused epilogue -> S = w2*(1-tanh(Z+b1)^2) in hi/lo bf16
__global__ __launch_bounds__(256)
void k_gemm1(const __bf16* __restrict__ Phi, const __bf16* __restrict__ Plo,
             const __bf16* __restrict__ Whi, const __bf16* __restrict__ Wlo,
             const float* __restrict__ bias, const float* __restrict__ w2,
             __bf16* __restrict__ Shi, __bf16* __restrict__ Slo)
{
    const int lane = threadIdx.x & 31;
    const int wv   = threadIdx.x >> 5;
    const int m0w  = blockIdx.x*128 + (wv & 3)*32;
    const int n0w  = blockIdx.y*64  + (wv >> 2)*32;
    const int lr   = lane & 15;
    const int half = lane >> 4;

    f32x8 acc00 = {}, acc01 = {}, acc10 = {}, acc11 = {};
    const size_t rowA0 = (size_t)(m0w + lr)      * T_PAD;
    const size_t rowA1 = (size_t)(m0w + 16 + lr) * T_PAD;
    const size_t colB0 = (size_t)(n0w + lr)      * 32 + 8*half;
    const size_t colB1 = (size_t)(n0w + 16 + lr) * 32 + 8*half;

    for (int kb = 0; kb < KB1; ++kb) {
        const int    k0 = kb*32;
        const size_t a0 = rowA0 + k0 + 8*half;
        const size_t a1 = rowA1 + k0 + 8*half;
        bf16x16 ah0 = ld_frag(Phi + a0, Phi + a0 + 16);
        bf16x16 ah1 = ld_frag(Phi + a1, Phi + a1 + 16);
        bf16x16 al0 = ld_frag(Plo + a0, Plo + a0 + 16);
        bf16x16 al1 = ld_frag(Plo + a1, Plo + a1 + 16);
        const size_t bb = (size_t)kb*8192;
        bf16x16 bh0 = ld_frag(Whi + bb + colB0, Whi + bb + colB0 + 16);
        bf16x16 bh1 = ld_frag(Whi + bb + colB1, Whi + bb + colB1 + 16);
        bf16x16 bl0 = ld_frag(Wlo + bb + colB0, Wlo + bb + colB0 + 16);
        bf16x16 bl1 = ld_frag(Wlo + bb + colB1, Wlo + bb + colB1 + 16);

        acc00 = WMMA_BF16(ah0, bh0, acc00);
        acc00 = WMMA_BF16(ah0, bl0, acc00);
        acc00 = WMMA_BF16(al0, bh0, acc00);
        acc01 = WMMA_BF16(ah0, bh1, acc01);
        acc01 = WMMA_BF16(ah0, bl1, acc01);
        acc01 = WMMA_BF16(al0, bh1, acc01);
        acc10 = WMMA_BF16(ah1, bh0, acc10);
        acc10 = WMMA_BF16(ah1, bl0, acc10);
        acc10 = WMMA_BF16(al1, bh0, acc10);
        acc11 = WMMA_BF16(ah1, bh1, acc11);
        acc11 = WMMA_BF16(ah1, bl1, acc11);
        acc11 = WMMA_BF16(al1, bh1, acc11);
    }

    auto epi = [&](const f32x8& c, int fm, int fn) {
        const int h  = n0w + fn*16 + lr;
        const float bh = bias[h], wh = w2[h];
        const int mb = m0w + fm*16 + half*8;
#pragma unroll
        for (int v = 0; v < 8; ++v) {
            float z  = c[v] + bh;
            float th = tanhf(z);
            float s  = wh * (1.f - th*th);
            __bf16 hi = (__bf16)s;
            __bf16 lo = (__bf16)(s - (float)hi);
            const size_t o = (size_t)(mb + v)*H_DIM + h;
            Shi[o] = hi;
            Slo[o] = lo;
        }
    };
    epi(acc00, 0, 0); epi(acc01, 0, 1); epi(acc10, 1, 0); epi(acc11, 1, 1);
}

// GEMM2: G = S @ W1^T (bf16x3), f32 output
__global__ __launch_bounds__(256)
void k_gemm2(const __bf16* __restrict__ Shi, const __bf16* __restrict__ Slo,
             const __bf16* __restrict__ WThi, const __bf16* __restrict__ WTlo,
             float* __restrict__ G)
{
    const int lane = threadIdx.x & 31;
    const int wv   = threadIdx.x >> 5;
    const int m0w  = blockIdx.x*128 + (wv & 3)*32;
    const int n0w  = blockIdx.y*64  + (wv >> 2)*32;   // t dimension
    const int lr   = lane & 15;
    const int half = lane >> 4;

    f32x8 acc00 = {}, acc01 = {}, acc10 = {}, acc11 = {};
    const size_t rowA0 = (size_t)(m0w + lr)      * H_DIM;
    const size_t rowA1 = (size_t)(m0w + 16 + lr) * H_DIM;
    const size_t colB0 = (size_t)(n0w + lr)      * 32 + 8*half;
    const size_t colB1 = (size_t)(n0w + 16 + lr) * 32 + 8*half;

#pragma unroll
    for (int kb = 0; kb < KB2; ++kb) {
        const int    k0 = kb*32;
        const size_t a0 = rowA0 + k0 + 8*half;
        const size_t a1 = rowA1 + k0 + 8*half;
        bf16x16 ah0 = ld_frag(Shi + a0, Shi + a0 + 16);
        bf16x16 ah1 = ld_frag(Shi + a1, Shi + a1 + 16);
        bf16x16 al0 = ld_frag(Slo + a0, Slo + a0 + 16);
        bf16x16 al1 = ld_frag(Slo + a1, Slo + a1 + 16);
        const size_t bb = (size_t)kb * (T_PAD*32);
        bf16x16 bh0 = ld_frag(WThi + bb + colB0, WThi + bb + colB0 + 16);
        bf16x16 bh1 = ld_frag(WThi + bb + colB1, WThi + bb + colB1 + 16);
        bf16x16 bl0 = ld_frag(WTlo + bb + colB0, WTlo + bb + colB0 + 16);
        bf16x16 bl1 = ld_frag(WTlo + bb + colB1, WTlo + bb + colB1 + 16);

        acc00 = WMMA_BF16(ah0, bh0, acc00);
        acc00 = WMMA_BF16(ah0, bl0, acc00);
        acc00 = WMMA_BF16(al0, bh0, acc00);
        acc01 = WMMA_BF16(ah0, bh1, acc01);
        acc01 = WMMA_BF16(ah0, bl1, acc01);
        acc01 = WMMA_BF16(al0, bh1, acc01);
        acc10 = WMMA_BF16(ah1, bh0, acc10);
        acc10 = WMMA_BF16(ah1, bl0, acc10);
        acc10 = WMMA_BF16(al1, bh0, acc10);
        acc11 = WMMA_BF16(ah1, bh1, acc11);
        acc11 = WMMA_BF16(ah1, bl1, acc11);
        acc11 = WMMA_BF16(al1, bh1, acc11);
    }

    auto sto = [&](const f32x8& c, int fm, int fn) {
        const int t  = n0w + fn*16 + lr;
        const int mb = m0w + fm*16 + half*8;
#pragma unroll
        for (int v = 0; v < 8; ++v)
            G[(size_t)(mb + v)*T_PAD + t] = c[v];
    };
    sto(acc00, 0, 0); sto(acc01, 0, 1); sto(acc10, 1, 0); sto(acc11, 1, 1);
}

// ---------------------------------------------------------------------------
// Stage 3: assemble out[b,t,i] = 0.5*(G(k)+G(k+1)) * (val(k)-val(k+1)), k=freq_index[i]
// ---------------------------------------------------------------------------
static __device__ __forceinline__ void fetch_vg(int q, int b, int t,
                                                float xv, float Gx, float g0t,
                                                const float* __restrict__ bs,
                                                const float* __restrict__ G,
                                                float& v, float& g)
{
    if (q <= 2)       { v = xv; g = Gx; }
    else if (q <= 99) {
        const size_t r = (size_t)((q - 3)*NB + b);
        v = bs[r*T_LEN + t];
        g = G[(r + NB)*T_PAD + t];
    } else            { v = 0.f; g = g0t; }
}

__global__ __launch_bounds__(256)
void k_assemble(const float* __restrict__ x, const float* __restrict__ bs,
                const float* __restrict__ G, const float* __restrict__ g0,
                const int* __restrict__ fidx, float* __restrict__ out)
{
    const int b = blockIdx.y;
    const int t = blockIdx.x*256 + threadIdx.x;
    if (t >= T_LEN) return;
    const float xv  = x[(size_t)b*T_LEN + t];
    const float Gx  = G[(size_t)b*T_PAD + t];
    const float g0t = g0[t];
    float* o = out + ((size_t)b*T_LEN + t) * 100;
    for (int i = 0; i < 100; ++i) {
        const int k = fidx[i];
        float vp, gp, vn, gn;
        fetch_vg(k,     b, t, xv, Gx, g0t, bs, G, vp, gp);
        fetch_vg(k + 1, b, t, xv, Gx, g0t, bs, G, vn, gn);
        o[i] = 0.5f * (gp + gn) * (vp - vn);
    }
}

// ---------------------------------------------------------------------------
extern "C" void kernel_launch(void* const* d_in, const int* in_sizes, int n_in,
                              void* d_out, int out_size, void* d_ws, size_t ws_size,
                              hipStream_t stream)
{
    const float* x      = (const float*)d_in[0];
    const float* b_all  = (const float*)d_in[1];
    const float* a_all  = (const float*)d_in[2];
    const float* zi_all = (const float*)d_in[3];
    const float* W1     = (const float*)d_in[4];
    const float* bias   = (const float*)d_in[5];
    const float* w2     = (const float*)d_in[6];
    const int*   fidx   = (const int*)d_in[7];
    float* out = (float*)d_out;

    char* wsp = (char*)d_ws;
    size_t off = 0;
    auto wsalloc = [&](size_t bytes) -> char* {
        char* p = wsp + off;
        off += (bytes + 255) & ~(size_t)255;
        return p;
    };
    float*  bs    = (float*) wsalloc((size_t)NF*NB*T_LEN*4);
    float*  yextG = (float*) wsalloc((size_t)NF*NB*YSTRIDE*4);  // reused as G (6272x3008 fits)
    float*  g0    = (float*) wsalloc((size_t)T_PAD*4);
    __bf16* Phi   = (__bf16*)wsalloc((size_t)M_ROWS*T_PAD*2);
    __bf16* Plo   = (__bf16*)wsalloc((size_t)M_ROWS*T_PAD*2);
    __bf16* Whi   = (__bf16*)wsalloc((size_t)KB1*8192*2);
    __bf16* Wlo   = (__bf16*)wsalloc((size_t)KB1*8192*2);
    __bf16* WThi  = (__bf16*)wsalloc((size_t)KB2*T_PAD*32*2);
    __bf16* WTlo  = (__bf16*)wsalloc((size_t)KB2*T_PAD*32*2);
    __bf16* Shi   = (__bf16*)wsalloc((size_t)M_ROWS*H_DIM*2);
    __bf16* Slo   = (__bf16*)wsalloc((size_t)M_ROWS*H_DIM*2);
    float*  G     = yextG;

    k_filtfilt<<<NF, NB, 0, stream>>>(x, b_all, a_all, zi_all, yextG, bs);
    k_packP  <<<dim3(12, M_ROWS), 256, 0, stream>>>(x, bs, Phi, Plo);
    k_packW  <<<(KB1*8192 + 255)/256, 256, 0, stream>>>(W1, Whi, Wlo);
    k_packWT <<<(KB2*T_PAD*32 + 255)/256, 256, 0, stream>>>(W1, WThi, WTlo);
    k_g0     <<<(T_PAD + 255)/256, 256, 0, stream>>>(W1, bias, w2, g0);
    k_gemm1  <<<dim3(M_ROWS/128, H_DIM/64), 256, 0, stream>>>(Phi, Plo, Whi, Wlo, bias, w2, Shi, Slo);
    k_gemm2  <<<dim3(M_ROWS/128, 47), 256, 0, stream>>>(Shi, Slo, WThi, WTlo, G);
    k_assemble<<<dim3(12, NB), 256, 0, stream>>>(x, bs, G, g0, fidx, out);

    (void)in_sizes; (void)n_in; (void)out_size; (void)ws_size;
}
